// SNNLayer_8229157339887
// MI455X (gfx1250) — matrix operand order repaired and checked by reading, not compiled, not run on hardware
//
#include <hip/hip_runtime.h>
#include <hip/hip_bf16.h>

typedef __bf16 bf16;
typedef __attribute__((ext_vector_type(16))) __bf16 v16bf;
typedef __attribute__((ext_vector_type(8)))  __bf16 v8bf;
typedef __attribute__((ext_vector_type(8)))  float  v8f;

#define NB 32          // batch
#define NT 32          // time steps
#define NN 4096        // oscillators
#define ND 16          // osc dim
#define NC 10          // classes
#define CLS_START_IDX 15
#define POOL_STEPS 17.0f

// ---------------------------------------------------------------------------
// 1) drive[b,t,n] = sum_d gamma[b,t,n,d] * softplus(input_w[d])
//    Streams the 268MB gamma tensor exactly once.
// ---------------------------------------------------------------------------
__global__ void drive_kernel(const float* __restrict__ gamma,
                             const float* __restrict__ input_w,
                             float* __restrict__ drive) {
    __shared__ float pw[ND];
    if (threadIdx.x < ND) {
        float x = input_w[threadIdx.x];
        // numerically stable softplus
        pw[threadIdx.x] = fmaxf(x, 0.0f) + log1pf(__expf(-fabsf(x)));
    }
    __syncthreads();
    size_t idx = (size_t)blockIdx.x * blockDim.x + threadIdx.x;  // over B*T*N
    const float4* g = reinterpret_cast<const float4*>(gamma + idx * ND);
    float acc = 0.0f;
    #pragma unroll
    for (int j = 0; j < 4; ++j) {
        float4 v = g[j];
        acc += v.x * pw[4 * j + 0] + v.y * pw[4 * j + 1] +
               v.z * pw[4 * j + 2] + v.w * pw[4 * j + 3];
    }
    drive[idx] = acc;
}

// ---------------------------------------------------------------------------
// 2) recT[n][k] = bf16(rec_w[k][n])  -- transpose + convert so WMMA B-fragment
//    loads are contiguous (K-major per column). 32x32 LDS tiles.
// ---------------------------------------------------------------------------
__global__ void transpose_convert_kernel(const float* __restrict__ rec_w,
                                         bf16* __restrict__ recT) {
    __shared__ float tile[32][33];
    const int nb = blockIdx.x * 32;   // n base (fast dim of input)
    const int kb = blockIdx.y * 32;   // k base (row dim of input)
    const int tx = threadIdx.x;       // 0..31
    const int ty = threadIdx.y;       // 0..7
    #pragma unroll
    for (int j = 0; j < 4; ++j) {
        int k = kb + ty + j * 8;
        tile[ty + j * 8][tx] = rec_w[(size_t)k * NN + nb + tx];
    }
    __syncthreads();
    #pragma unroll
    for (int j = 0; j < 4; ++j) {
        int n = nb + ty + j * 8;
        recT[(size_t)n * NN + kb + tx] = (bf16)tile[tx][ty + j * 8];
    }
}

// ---------------------------------------------------------------------------
// 3) zero-init recurrent state
// ---------------------------------------------------------------------------
__global__ void init_state_kernel(float* m_state, float* s_f32, float* pooled,
                                  bf16* s_bf0, bf16* s_bf1) {
    int i = blockIdx.x * blockDim.x + threadIdx.x;
    if (i < NB * NN) {
        m_state[i] = 0.0f;
        s_f32[i]   = 0.0f;
        pooled[i]  = 0.0f;
        s_bf0[i]   = (bf16)0.0f;
        s_bf1[i]   = (bf16)0.0f;
    }
}

// ---------------------------------------------------------------------------
// 4) one recurrent step:  rec = 0.1 * (s_prev @ rec_w)  via bf16 WMMA,
//    fused with the pointwise membrane update + sigmoid + pooled accumulation.
//    One wave owns one 16x16 M-tile x TWO adjacent 16-wide N-tiles: the A
//    fragment is loaded once per K-chunk and reused by both WMMAs
//    (2 A-loads + 4 B-loads -> 2 wmma, i.e. 3 VMEM per WMMA instead of 4).
//    Grid: 64 blocks x 128 threads (4 waves) -> 2 M-tiles x 128 N-pairs.
// ---------------------------------------------------------------------------
__global__ void __launch_bounds__(128)
step_kernel(const bf16* __restrict__ recT,
            const float* __restrict__ drive,
            float* __restrict__ m_state,
            float* __restrict__ s_f32,
            const bf16* __restrict__ s_in,
            bf16* __restrict__ s_out,
            float* __restrict__ pooled,
            int t) {
    const int lane   = threadIdx.x & 31;
    const int wave   = threadIdx.x >> 5;              // 0..3
    const int m_tile = blockIdx.x >> 5;               // 0..1
    const int n_pair = (blockIdx.x & 31) * 4 + wave;  // 0..127
    const int n_tile0 = n_pair * 2;                   // two adjacent N-tiles
    const int n_tile1 = n_pair * 2 + 1;

    const int col   = lane & 15;
    const int khalf = lane >> 4;                      // 0 or 1

    // A fragment source: row m of s_prev (bf16), contiguous in K.
    //   lane<16 : M=lane,    K = k0+0..15
    //   lane>=16: M=lane-16, K = k0+16..31
    const bf16* arow  = s_in + (size_t)(m_tile * 16 + col) * NN;
    // B fragment source: column n of rec_w == row n of recT, contiguous in K.
    //   lane<16 : N=lane,    K = k0+0..15
    //   lane>=16: N=lane-16, K = k0+16..31
    const bf16* brow0 = recT + (size_t)(n_tile0 * 16 + col) * NN;
    const bf16* brow1 = recT + (size_t)(n_tile1 * 16 + col) * NN;

    v8f c0 = {};
    v8f c1 = {};
    for (int k0 = 0; k0 < NN; k0 += 32) {
        v8bf alo  = *reinterpret_cast<const v8bf*>(arow + k0 + khalf * 16);
        v8bf ahi  = *reinterpret_cast<const v8bf*>(arow + k0 + khalf * 16 + 8);
        v8bf b0lo = *reinterpret_cast<const v8bf*>(brow0 + k0 + khalf * 16);
        v8bf b0hi = *reinterpret_cast<const v8bf*>(brow0 + k0 + khalf * 16 + 8);
        v8bf b1lo = *reinterpret_cast<const v8bf*>(brow1 + k0 + khalf * 16);
        v8bf b1hi = *reinterpret_cast<const v8bf*>(brow1 + k0 + khalf * 16 + 8);
        v16bf a  = __builtin_shufflevector(alo, ahi, 0, 1, 2, 3, 4, 5, 6, 7,
                                           8, 9, 10, 11, 12, 13, 14, 15);
        v16bf b0 = __builtin_shufflevector(b0lo, b0hi, 0, 1, 2, 3, 4, 5, 6, 7,
                                           8, 9, 10, 11, 12, 13, 14, 15);
        v16bf b1 = __builtin_shufflevector(b1lo, b1hi, 0, 1, 2, 3, 4, 5, 6, 7,
                                           8, 9, 10, 11, 12, 13, 14, 15);
        c0 = __builtin_amdgcn_wmma_f32_16x16x32_bf16(
                 false, a, false, b0, (short)0, c0, false, false);
        c1 = __builtin_amdgcn_wmma_f32_16x16x32_bf16(
                 false, a, false, b1, (short)0, c1, false, false);
    }

    // Epilogue. C/D layout: lane<16 -> (M=r, N=lane); lane>=16 -> (M=8+r, N=lane-16)
    #pragma unroll
    for (int j = 0; j < 2; ++j) {
        const int n = (n_tile0 + j) * 16 + col;
        #pragma unroll
        for (int r = 0; r < 8; ++r) {
            const int b = m_tile * 16 + khalf * 8 + r;    // batch index
            const size_t idx = (size_t)b * NN + n;
            float rec = 0.1f * (j == 0 ? c0[r] : c1[r]);
            float mp  = m_state[idx];
            float sp  = s_f32[idx];
            float dr  = drive[((size_t)b * NT + t) * NN + n];
            float mn  = 0.9f * mp + dr + rec - 0.5f * sp;
            float sn  = 1.0f / (1.0f + __expf(-8.0f * (mn - 0.5f)));
            m_state[idx] = mn;
            s_f32[idx]   = sn;
            s_out[idx]   = (bf16)sn;
            if (t >= CLS_START_IDX) pooled[idx] += sn;
        }
    }
}

// ---------------------------------------------------------------------------
// 5) pooled mean -> layernorm over N -> logits. One block per batch row.
// ---------------------------------------------------------------------------
__global__ void finalize_kernel(const float* __restrict__ pooled,
                                const float* __restrict__ ln_g,
                                const float* __restrict__ ln_b,
                                const float* __restrict__ cls_w,
                                const float* __restrict__ cls_b,
                                float* __restrict__ out) {
    const int b   = blockIdx.x;
    const int tid = threadIdx.x;     // 256 threads
    __shared__ float red[256];

    float sum = 0.0f, sq = 0.0f;
    for (int n = tid; n < NN; n += 256) {
        float p = pooled[(size_t)b * NN + n] * (1.0f / POOL_STEPS);
        sum += p;
        sq  += p * p;
    }
    red[tid] = sum; __syncthreads();
    for (int s = 128; s > 0; s >>= 1) {
        if (tid < s) red[tid] += red[tid + s];
        __syncthreads();
    }
    float mu = red[0] * (1.0f / NN);
    __syncthreads();
    red[tid] = sq; __syncthreads();
    for (int s = 128; s > 0; s >>= 1) {
        if (tid < s) red[tid] += red[tid + s];
        __syncthreads();
    }
    float var  = red[0] * (1.0f / NN) - mu * mu;
    __syncthreads();
    float rstd = rsqrtf(var + 1e-5f);

    float acc[NC];
    #pragma unroll
    for (int c = 0; c < NC; ++c) acc[c] = 0.0f;
    for (int n = tid; n < NN; n += 256) {
        float p  = pooled[(size_t)b * NN + n] * (1.0f / POOL_STEPS);
        float nv = (p - mu) * rstd * ln_g[n] + ln_b[n];
        #pragma unroll
        for (int c = 0; c < NC; ++c) acc[c] += nv * cls_w[(size_t)c * NN + n];
    }
    #pragma unroll
    for (int c = 0; c < NC; ++c) {
        red[tid] = acc[c]; __syncthreads();
        for (int s = 128; s > 0; s >>= 1) {
            if (tid < s) red[tid] += red[tid + s];
            __syncthreads();
        }
        if (tid == 0) out[b * NC + c] = red[0] + cls_b[c];
        __syncthreads();
    }
}

// ---------------------------------------------------------------------------
extern "C" void kernel_launch(void* const* d_in, const int* in_sizes, int n_in,
                              void* d_out, int out_size, void* d_ws, size_t ws_size,
                              hipStream_t stream) {
    const float* gamma   = (const float*)d_in[0];  // [B,T,N,D]
    const float* input_w = (const float*)d_in[1];  // [1,D]
    const float* rec_w   = (const float*)d_in[2];  // [N,N]
    const float* ln_g    = (const float*)d_in[3];  // [N]
    const float* ln_b    = (const float*)d_in[4];  // [N]
    const float* cls_w   = (const float*)d_in[5];  // [C,N]
    const float* cls_b   = (const float*)d_in[6];  // [C]
    float*       out     = (float*)d_out;          // [B,C]

    // workspace carve-up (~52.4 MB total)
    char* ws = (char*)d_ws;
    bf16*  recT    = (bf16*)ws;  ws += (size_t)NN * NN * sizeof(bf16);       // 33.5 MB
    float* drive   = (float*)ws; ws += (size_t)NB * NT * NN * sizeof(float); // 16.8 MB
    float* m_state = (float*)ws; ws += (size_t)NB * NN * sizeof(float);
    float* s_f32   = (float*)ws; ws += (size_t)NB * NN * sizeof(float);
    bf16*  s_bf0   = (bf16*)ws;  ws += (size_t)NB * NN * sizeof(bf16);
    bf16*  s_bf1   = (bf16*)ws;  ws += (size_t)NB * NN * sizeof(bf16);
    float* pooled  = (float*)ws; ws += (size_t)NB * NN * sizeof(float);

    // 1) synaptic drive (streams gamma once)
    drive_kernel<<<(NB * NT * NN) / 256, 256, 0, stream>>>(gamma, input_w, drive);

    // 2) rec_w -> transposed bf16 (stays resident in 192MB L2 for all 32 steps)
    transpose_convert_kernel<<<dim3(NN / 32, NN / 32), dim3(32, 8), 0, stream>>>(rec_w, recT);

    // 3) zero state
    init_state_kernel<<<(NB * NN + 255) / 256, 256, 0, stream>>>(m_state, s_f32, pooled,
                                                                 s_bf0, s_bf1);

    // 4) recurrent scan: one launch per step (grid-wide dependency each step)
    for (int t = 0; t < NT; ++t) {
        const bf16* sin  = (t & 1) ? s_bf1 : s_bf0;
        bf16*       sout = (t & 1) ? s_bf0 : s_bf1;
        step_kernel<<<64, 128, 0, stream>>>(recT, drive, m_state, s_f32,
                                            sin, sout, pooled, t);
    }

    // 5) pooled mean -> layernorm -> classifier
    finalize_kernel<<<NB, 256, 0, stream>>>(pooled, ln_g, ln_b, cls_w, cls_b, out);
}
